// Model_22703197127028
// MI455X (gfx1250) — matrix operand order repaired and checked by reading, not compiled, not run on hardware
//
#include <hip/hip_runtime.h>

// ---------------------------------------------------------------------------
// TextLevelGNN forward for MI455X (gfx1250, wave32).
//
//   h      = node_emb[node_ids]                    [N, D]
//   msg    = h[edge_src] * edge_w[edge_ids]        [E, D]
//   h_max  = segment_max(msg, edge_dst)            [N, D]  (-inf empty -> 0)
//   pooled = segment_sum(h_max, node_seg)          [B, D]
//   out    = relu(pooled) @ W + b                  [B, C]
//
// One workgroup per graph: the graph's 200xD f32 node tile (160,000 B) plus
// its 200xD max accumulator (160,000 B) exactly fill the CDNA5 WGP's 320 KB
// LDS. The tile is DMA'd in with GLOBAL_LOAD_ASYNC_TO_LDS_B128 (ASYNCcnt);
// scatter-max runs on native LDS float-max atomics (ds_max_num_f32). The
// head GEMM runs on v_wmma_f32_16x16x32_f16.
// ---------------------------------------------------------------------------

typedef __attribute__((ext_vector_type(16))) _Float16 v16h;
typedef __attribute__((ext_vector_type(8)))  _Float16 v8h;
typedef __attribute__((ext_vector_type(8)))  float    v8f;

constexpr int B_  = 256;    // graphs
constexpr int NPG = 200;    // nodes per graph
constexpr int EPG = 1600;   // edges per graph
constexpr int D_  = 200;    // hidden size
constexpr int C_  = 20;     // classes
constexpr int KP  = 224;    // D padded to multiple of 32 (WMMA K)
constexpr int NP  = 32;     // C padded to two 16-wide WMMA tiles

#if defined(__HIP_DEVICE_COMPILE__) && \
    __has_builtin(__builtin_amdgcn_global_load_async_to_lds_b128)
#define USE_ASYNC_LDS 1
// Builtin signature (from hipcc diagnostic): param0 is int4(vector_size 16)
// pointer in AS1 (global), param1 the same in AS3 (LDS), then imm offset, cpol.
typedef int v4i_vs __attribute__((vector_size(16)));
typedef __attribute__((address_space(1))) v4i_vs* as1_b128_t;
typedef __attribute__((address_space(3))) v4i_vs* as3_b128_t;
#else
#define USE_ASYNC_LDS 0
#endif

// ---------------------------------------------------------------------------
// Kernel 1: per-graph gather -> LDS, edge scatter-max in LDS, pool + ReLU,
// store pooled row as f16 (zero-padded to KP) for the WMMA head.
// ---------------------------------------------------------------------------
__global__ __launch_bounds__(256)
void graph_max_pool_kernel(const float* __restrict__ node_emb,
                           const float* __restrict__ edge_w,
                           const int*   __restrict__ node_ids,
                           const int*   __restrict__ edge_src,
                           const int*   __restrict__ edge_dst,
                           const int*   __restrict__ edge_ids,
                           _Float16*    __restrict__ poolH)   // [B_][KP] f16
{
    extern __shared__ float smem[];
    float* sh   = smem;             // [NPG][D_]  node hidden states
    float* smax = smem + NPG * D_;  // [NPG][D_]  running max per dst node

    const int g = blockIdx.x;
    const int t = threadIdx.x;

    // ---- Phase 1: gather this graph's node rows into LDS (16 B / lane) ----
    constexpr int NV = D_ / 4;                  // 50 float4 per row
    for (int idx = t; idx < NPG * NV; idx += blockDim.x) {
        const int i = idx / NV, v = idx % NV;
        const int row = node_ids[g * NPG + i];
        const float* gsrc = node_emb + (size_t)row * D_ + v * 4;
        float*       ldst = sh + i * D_ + v * 4;
#if USE_ASYNC_LDS
        __builtin_amdgcn_global_load_async_to_lds_b128(
            (as1_b128_t)gsrc, (as3_b128_t)ldst, /*offset=*/0, /*cpol=*/0);
#else
        *(float4*)ldst = *(const float4*)gsrc;
#endif
    }
    // -inf init of the max tile overlaps with the async DMA above.
    for (int idx = t; idx < NPG * D_; idx += blockDim.x)
        smax[idx] = -__builtin_inff();
#if USE_ASYNC_LDS
#if __has_builtin(__builtin_amdgcn_s_wait_asynccnt)
    __builtin_amdgcn_s_wait_asynccnt(0);
#else
    asm volatile("s_wait_asynccnt 0" ::: "memory");
#endif
#endif
    __syncthreads();

    // ---- Phase 2: edges. One edge per wave; 32 lanes sweep D. -------------
    const int wave = t >> 5, lane = t & 31;
    const int ebase = g * EPG;
    for (int e = wave; e < EPG; e += 8) {
        const int   src = edge_src[ebase + e] - g * NPG;   // graph-local
        const int   dst = edge_dst[ebase + e] - g * NPG;
        const float w   = edge_w[edge_ids[ebase + e]];
        const float* hs = sh   + src * D_;
        float*       hm = smax + dst * D_;
        for (int d = lane; d < D_; d += 32) {
            __hip_atomic_fetch_max(hm + d, hs[d] * w,
                                   __ATOMIC_RELAXED, __HIP_MEMORY_SCOPE_WORKGROUP);
        }
    }
    __syncthreads();

    // ---- Phase 3: per-dim sum over nodes (empty segs -> 0), ReLU, f16 -----
    if (t < KP) {
        float acc = 0.0f;
        if (t < D_) {
            for (int i = 0; i < NPG; ++i) {
                const float m = smax[i * D_ + t];
                acc += (m == -__builtin_inff()) ? 0.0f : m;
            }
            acc = fmaxf(acc, 0.0f);            // relu(pooled)
        }
        poolH[(size_t)g * KP + t] = (_Float16)acc;   // K in [D_,KP) padded 0
    }
}

// ---------------------------------------------------------------------------
// Kernel 2: pad/convert/transpose W[D_,C_] f32 -> Wht[NP][KP] f16, so the
// WMMA B-fragment reads two 16-byte K-contiguous chunks per lane.
// ---------------------------------------------------------------------------
__global__ __launch_bounds__(256)
void wconv_kernel(const float* __restrict__ W, _Float16* __restrict__ Wht)
{
    for (int idx = threadIdx.x + blockIdx.x * blockDim.x; idx < NP * KP;
         idx += blockDim.x * gridDim.x) {
        const int n = idx / KP, k = idx % KP;
        Wht[idx] = (k < D_ && n < C_) ? (_Float16)W[k * C_ + n] : (_Float16)0.0f;
    }
}

// ---------------------------------------------------------------------------
// Kernel 3: out[256,20] = poolH[256,KP] @ W + b via WMMA f16->f32.
// One wave per 16x16 output tile; 16 M-tiles x 2 N-tiles = 32 waves.
// Fragment layouts per CDNA5 ISA 7.12.2 (wave32):
//   A 16x32: lanes 0-15 hold M=lane, halves j<8 -> K=j, j>=8 -> K=j+8;
//            lanes 16-31 same M, K shifted by +8.
//   B 32x16: lanes 0-15 hold N=lane, same K mapping as A (lanes 16-31 +8).
//   C/D 16x16 f32: VGPR r -> M = r + (lane<16 ? 0 : 8), N = lane&15.
// ---------------------------------------------------------------------------
__global__ __launch_bounds__(256)
void head_wmma_kernel(const _Float16* __restrict__ poolH,
                      const _Float16* __restrict__ Wht,   // [NP][KP]
                      const float*    __restrict__ bias,
                      float*          __restrict__ out)
{
    const int lane  = threadIdx.x & 31;
    const int gw    = blockIdx.x * (blockDim.x >> 5) + (threadIdx.x >> 5);
    const int mtile = gw >> 1;          // 0..15
    const int ntile = gw & 1;           // 0..1
    const int hi    = lane >> 4;        // K-half selector
    const int mn    = lane & 15;

    v8f acc = {};
    const int row = mtile * 16 + mn;    // A-row this lane supplies
    const int n   = ntile * 16 + mn;    // B-column this lane supplies

    #pragma unroll
    for (int kc = 0; kc < KP / 32; ++kc) {
        // A fragment: two aligned 8-half chunks at K = kc*32 + hi*8 (+16)
        const _Float16* pA = poolH + (size_t)row * KP + kc * 32 + hi * 8;
        const v8h alo = *(const v8h*)(pA);
        const v8h ahi = *(const v8h*)(pA + 16);
        v16h a;
        #pragma unroll
        for (int j = 0; j < 8; ++j) { a[j] = alo[j]; a[j + 8] = ahi[j]; }

        // B fragment: two aligned 8-half chunks at K = kc*32 + hi*8 (+16) (transposed W)
        const _Float16* pB = Wht + (size_t)n * KP + kc * 32 + hi * 8;
        const v8h blo = *(const v8h*)(pB);
        const v8h bhi = *(const v8h*)(pB + 16);
        v16h b;
        #pragma unroll
        for (int j = 0; j < 8; ++j) { b[j] = blo[j]; b[j + 8] = bhi[j]; }

        acc = __builtin_amdgcn_wmma_f32_16x16x32_f16(
                  false, a, false, b, (short)0, acc, false, false);
    }

    const int col = ntile * 16 + mn;
    if (col < C_) {
        const float bb = bias[col];
        #pragma unroll
        for (int r = 0; r < 8; ++r) {
            const int m = mtile * 16 + r + (hi ? 8 : 0);
            out[m * C_ + col] = acc[r] + bb;
        }
    }
}

// ---------------------------------------------------------------------------
// Launch. Inputs (setup_inputs order):
//   0 node_emb f32[V*D], 1 edge_w f32[ENUM], 2 W f32[D*C], 3 b f32[C],
//   4 node_ids i32[N], 5 edge_src i32[E], 6 edge_dst i32[E],
//   7 edge_ids i32[E], 8 node_seg i32[N] (unused: segments are contiguous)
// ---------------------------------------------------------------------------
extern "C" void kernel_launch(void* const* d_in, const int* in_sizes, int n_in,
                              void* d_out, int out_size, void* d_ws, size_t ws_size,
                              hipStream_t stream)
{
    const float* node_emb = (const float*)d_in[0];
    const float* edge_w   = (const float*)d_in[1];
    const float* W        = (const float*)d_in[2];
    const float* bias     = (const float*)d_in[3];
    const int*   node_ids = (const int*)d_in[4];
    const int*   edge_src = (const int*)d_in[5];
    const int*   edge_dst = (const int*)d_in[6];
    const int*   edge_ids = (const int*)d_in[7];
    float*       out      = (float*)d_out;

    _Float16* poolH = (_Float16*)d_ws;                 // B_*KP f16 = 112 KB
    _Float16* Wht   = poolH + (size_t)B_ * KP;         // NP*KP f16 =  14 KB

    const size_t lds_bytes = 2u * NPG * D_ * sizeof(float);  // 320,000 B / WGP

    hipLaunchKernelGGL(graph_max_pool_kernel, dim3(B_), dim3(256), lds_bytes,
                       stream, node_emb, edge_w, node_ids, edge_src, edge_dst,
                       edge_ids, poolH);
    hipLaunchKernelGGL(wconv_kernel, dim3(4), dim3(256), 0, stream, W, Wht);
    hipLaunchKernelGGL(head_wmma_kernel, dim3(4), dim3(256), 0, stream,
                       poolH, Wht, bias, out);
}
